// StackedAttentionModel_59004260713074
// MI455X (gfx1250) — matrix operand order repaired and checked by reading, not compiled, not run on hardware
//
#include <hip/hip_runtime.h>
#include <hip/hip_bf16.h>
#include <math.h>

// ---------------------------------------------------------------------------
// Problem constants (from the reference)
// ---------------------------------------------------------------------------
static constexpr int Lc = 8;      // layers
static constexpr int Ec = 1024;   // embedding dim
static constexpr int Hc = 16;     // heads
static constexpr int Sc = 2048;   // sequence
static constexpr int Vc = 32000;  // vocab
static constexpr int Fc = 4096;   // ffn dim
static constexpr int Dc = 64;     // head dim

typedef __bf16 v16bf __attribute__((ext_vector_type(16)));
typedef __bf16 v8bf  __attribute__((ext_vector_type(8)));
typedef float  v8f   __attribute__((ext_vector_type(8)));

union BFrag { v16bf v; v8bf h[2]; };

// ---------------------------------------------------------------------------
// Embedding gather:  x[s, :] = emb_table[inpt[s], :]
// ---------------------------------------------------------------------------
__global__ __launch_bounds__(256) void embed_kernel(
    const int* __restrict__ inpt, const float* __restrict__ table,
    float* __restrict__ x)
{
    int i = blockIdx.x * blockDim.x + threadIdx.x;          // over S*E/4
    int s = i / (Ec / 4);
    int c = (i % (Ec / 4)) * 4;
    int tok = inpt[s];
    float4 v = *(const float4*)(table + (size_t)tok * Ec + c);
    *(float4*)(x + (size_t)s * Ec + c) = v;
}

// ---------------------------------------------------------------------------
// bf16 WMMA GEMM:  C[M,N] = act(A[M,K] @ B[K,N] + bias)
// fp32 global data converted to bf16 while staging into double-buffered LDS.
// Block tile 128x128x64, 256 threads = 8 waves (2x4), wave tile 64x32.
// ---------------------------------------------------------------------------
static constexpr int BM  = 128;
static constexpr int BN  = 128;
static constexpr int BK  = 64;
static constexpr int BKP = BK + 8;   // pad to de-conflict LDS banks

__global__ __launch_bounds__(256) void gemm_bf16_wmma(
    const float* __restrict__ A, const float* __restrict__ B,
    const float* __restrict__ bias, float* __restrict__ C,
    int M, int N, int K, int relu)
{
    __shared__ __align__(16) __bf16 sA[2][BM][BKP];   // [buf][m][k]
    __shared__ __align__(16) __bf16 sB[2][BN][BKP];   // [buf][n][k] (B^T)

    const int tid  = threadIdx.x;
    const int wave = tid >> 5;
    const int lane = tid & 31;
    const int hf   = lane >> 4;      // half-wave select
    const int l15  = lane & 15;

    const int bm = blockIdx.x * BM;
    const int bn = blockIdx.y * BN;
    const int wrow = (wave >> 2) * 64;   // 0 / 64
    const int wcol = (wave & 3) * 32;    // 0 / 32 / 64 / 96

    v8f acc[4][2] = {};
    float4 ra[8], rb[8];                 // staging registers (batched loads)

    const int nk = K / BK;

    // ---- batched global loads for one k-tile (fixed trip count) ----
    auto load_tiles = [&](int k0) {
        #pragma unroll
        for (int u = 0; u < 8; ++u) {
            int i = tid * 4 + u * 1024;          // A: 128x64 elems
            int r = i >> 6, c = i & 63;
            ra[u] = *(const float4*)(A + (size_t)(bm + r) * K + k0 + c);
        }
        #pragma unroll
        for (int u = 0; u < 8; ++u) {
            int i = tid * 4 + u * 1024;          // B: 64x128 elems
            int r = i >> 7, c = i & 127;
            rb[u] = *(const float4*)(B + (size_t)(k0 + r) * N + bn + c);
        }
    };
    // ---- convert + store staged registers into LDS buffer ----
    auto store_tiles = [&](int buf) {
        #pragma unroll
        for (int u = 0; u < 8; ++u) {
            int i = tid * 4 + u * 1024;
            int r = i >> 6, c = i & 63;
            __bf16* p = &sA[buf][r][c];
            p[0] = (__bf16)ra[u].x; p[1] = (__bf16)ra[u].y;
            p[2] = (__bf16)ra[u].z; p[3] = (__bf16)ra[u].w;
        }
        #pragma unroll
        for (int u = 0; u < 8; ++u) {
            int i = tid * 4 + u * 1024;
            int r = i >> 7, c = i & 127;
            sB[buf][c + 0][r] = (__bf16)rb[u].x;
            sB[buf][c + 1][r] = (__bf16)rb[u].y;
            sB[buf][c + 2][r] = (__bf16)rb[u].z;
            sB[buf][c + 3][r] = (__bf16)rb[u].w;
        }
    };

    load_tiles(0);
    for (int kt = 0; kt < nk; ++kt) {
        const int buf = kt & 1;
        store_tiles(buf);
        __syncthreads();                         // stores visible to block
        if (kt + 1 < nk) load_tiles((kt + 1) * BK);   // overlap with WMMA

        #pragma unroll
        for (int kk = 0; kk < 2; ++kk) {         // two K=32 sub-steps
            BFrag af[4], bfg[2];
            #pragma unroll
            for (int i = 0; i < 4; ++i) {
                int r = wrow + i * 16 + l15;
                af[i].h[0] = *(const v8bf*)&sA[buf][r][kk * 32 + hf * 8];
                af[i].h[1] = *(const v8bf*)&sA[buf][r][kk * 32 + 16 + hf * 8];
            }
            #pragma unroll
            for (int j = 0; j < 2; ++j) {
                int n = wcol + j * 16 + l15;
                bfg[j].h[0] = *(const v8bf*)&sB[buf][n][kk * 32 + hf * 8];
                bfg[j].h[1] = *(const v8bf*)&sB[buf][n][kk * 32 + 16 + hf * 8];
            }
            #pragma unroll
            for (int i = 0; i < 4; ++i)
                #pragma unroll
                for (int j = 0; j < 2; ++j)
                    acc[i][j] = __builtin_amdgcn_wmma_f32_16x16x32_bf16(
                        false, af[i].v, false, bfg[j].v, (short)0, acc[i][j],
                        false, false);
        }
        // single barrier per iteration is safe: a wave's fragment loads are
        // consumed (waited) before it can signal the next barrier, and the
        // next store goes to the other buffer.
    }

    // ---- epilogue: bias + relu + store (C layout: lane=N col, vgpr=row) ----
    #pragma unroll
    for (int i = 0; i < 4; ++i) {
        #pragma unroll
        for (int j = 0; j < 2; ++j) {
            int n = bn + wcol + j * 16 + l15;
            float bv = bias ? bias[n] : 0.0f;
            #pragma unroll
            for (int v = 0; v < 8; ++v) {
                int m = bm + wrow + i * 16 + hf * 8 + v;
                float val = acc[i][j][v] + bv;
                if (relu) val = fmaxf(val, 0.0f);
                C[(size_t)m * N + n] = val;
            }
        }
    }
}

// ---------------------------------------------------------------------------
// Flash-style causal attention. One block = one head x 128 query rows.
// 8 waves; each wave owns 16 query rows. Q fragment lives in registers,
// K/V key-blocks of 64 staged in LDS (bf16, V transposed). Online softmax.
// ---------------------------------------------------------------------------
__global__ __launch_bounds__(256) void attn_wmma(
    const float* __restrict__ Q, const float* __restrict__ Kg,
    const float* __restrict__ Vg, float* __restrict__ O)
{
    __shared__ __align__(16) __bf16 sK[64][Dc + 8];     // [key][d]
    __shared__ __align__(16) __bf16 sV[Dc][64 + 8];     // [d][key]
    __shared__ __align__(16) __bf16 sP[8][16][32 + 8];  // per-wave probs

    const int tid  = threadIdx.x;
    const int wave = tid >> 5;
    const int lane = tid & 31;
    const int hf   = lane >> 4;
    const int l15  = lane & 15;

    const int h    = blockIdx.y;
    const int hoff = h * Dc;
    const int q0   = blockIdx.x * 128;
    const int qrow0 = q0 + wave * 16;

    // ---- Q fragment (A layout) straight from global, f32 -> bf16 ----
    BFrag qf[2];
    {
        const int qr = qrow0 + l15;
        #pragma unroll
        for (int kk = 0; kk < 2; ++kk) {
            const float* qp = Q + (size_t)qr * Ec + hoff + kk * 32 + hf * 8;
            #pragma unroll
            for (int t = 0; t < 8; ++t) {
                qf[kk].h[0][t] = (__bf16)qp[t];
                qf[kk].h[1][t] = (__bf16)qp[t + 16];
            }
        }
    }

    v8f   o[4] = {};
    float mrow[8], lrow[8];
    #pragma unroll
    for (int v = 0; v < 8; ++v) { mrow[v] = -1e30f; lrow[v] = 0.0f; }

    const int kmax = q0 + 127;                 // causal upper bound for block
    for (int k0 = 0; k0 <= kmax; k0 += 64) {
        // ---- stage K (row-major) and V (transposed), batched loads ----
        float4 rk[4], rv[4];
        #pragma unroll
        for (int u = 0; u < 4; ++u) {
            int i = tid * 4 + u * 1024;        // 64x64 elems
            int r = i >> 6, c = i & 63;
            rk[u] = *(const float4*)(Kg + (size_t)(k0 + r) * Ec + hoff + c);
            rv[u] = *(const float4*)(Vg + (size_t)(k0 + r) * Ec + hoff + c);
        }
        #pragma unroll
        for (int u = 0; u < 4; ++u) {
            int i = tid * 4 + u * 1024;
            int r = i >> 6, c = i & 63;
            __bf16* p = &sK[r][c];
            p[0] = (__bf16)rk[u].x; p[1] = (__bf16)rk[u].y;
            p[2] = (__bf16)rk[u].z; p[3] = (__bf16)rk[u].w;
            sV[c + 0][r] = (__bf16)rv[u].x;
            sV[c + 1][r] = (__bf16)rv[u].y;
            sV[c + 2][r] = (__bf16)rv[u].z;
            sV[c + 3][r] = (__bf16)rv[u].w;
        }
        __syncthreads();

        for (int c32 = 0; c32 < 64; c32 += 32) {
            if (k0 + c32 <= qrow0 + 15) {      // chunk not fully masked
                // ---- scores: two 16-key tiles, K-dim 64 = 2 WMMA steps ----
                v8f sc[2] = {};
                #pragma unroll
                for (int kk = 0; kk < 2; ++kk) {
                    #pragma unroll
                    for (int t = 0; t < 2; ++t) {
                        BFrag kb;
                        int key = c32 + t * 16 + l15;
                        kb.h[0] = *(const v8bf*)&sK[key][kk * 32 + hf * 8];
                        kb.h[1] = *(const v8bf*)&sK[key][kk * 32 + 16 + hf * 8];
                        sc[t] = __builtin_amdgcn_wmma_f32_16x16x32_bf16(
                            false, qf[kk].v, false, kb.v, (short)0, sc[t],
                            false, false);
                    }
                }
                // ---- scale + causal mask ----
                const int qg = qrow0 + hf * 8;
                #pragma unroll
                for (int t = 0; t < 2; ++t)
                    #pragma unroll
                    for (int v = 0; v < 8; ++v) {
                        int kg = k0 + c32 + t * 16 + l15;
                        float s = sc[t][v] * 0.125f;          // 1/sqrt(64)
                        if (kg > qg + v) s = -1e9f;
                        sc[t][v] = s;
                    }
                // ---- online softmax (rows split: lanes 0-15 vs 16-31) ----
                float rmax[8], rsum[8];
                #pragma unroll
                for (int v = 0; v < 8; ++v)
                    rmax[v] = fmaxf(sc[0][v], sc[1][v]);
                #pragma unroll
                for (int off = 1; off < 16; off <<= 1)
                    #pragma unroll
                    for (int v = 0; v < 8; ++v)
                        rmax[v] = fmaxf(rmax[v], __shfl_xor(rmax[v], off, 32));
                float corr[8];
                #pragma unroll
                for (int v = 0; v < 8; ++v) {
                    float mn = fmaxf(mrow[v], rmax[v]);
                    corr[v]  = __expf(mrow[v] - mn);
                    mrow[v]  = mn;
                    rsum[v]  = 0.0f;
                }
                #pragma unroll
                for (int t = 0; t < 2; ++t)
                    #pragma unroll
                    for (int v = 0; v < 8; ++v) {
                        float p = __expf(sc[t][v] - mrow[v]);
                        sc[t][v] = p;
                        rsum[v] += p;
                    }
                #pragma unroll
                for (int off = 1; off < 16; off <<= 1)
                    #pragma unroll
                    for (int v = 0; v < 8; ++v)
                        rsum[v] += __shfl_xor(rsum[v], off, 32);
                #pragma unroll
                for (int v = 0; v < 8; ++v)
                    lrow[v] = lrow[v] * corr[v] + rsum[v];
                #pragma unroll
                for (int j = 0; j < 4; ++j)
                    #pragma unroll
                    for (int v = 0; v < 8; ++v)
                        o[j][v] *= corr[v];
                // ---- C-layout probs -> LDS -> A-layout fragment ----
                #pragma unroll
                for (int t = 0; t < 2; ++t)
                    #pragma unroll
                    for (int v = 0; v < 8; ++v)
                        sP[wave][hf * 8 + v][t * 16 + l15] = (__bf16)sc[t][v];
                BFrag pf;
                pf.h[0] = *(const v8bf*)&sP[wave][l15][hf * 8];
                pf.h[1] = *(const v8bf*)&sP[wave][l15][16 + hf * 8];
                // ---- O += P @ V  (4 d-tiles, K = 32 keys) ----
                #pragma unroll
                for (int j = 0; j < 4; ++j) {
                    BFrag vf;
                    int d = j * 16 + l15;
                    vf.h[0] = *(const v8bf*)&sV[d][c32 + hf * 8];
                    vf.h[1] = *(const v8bf*)&sV[d][c32 + 16 + hf * 8];
                    o[j] = __builtin_amdgcn_wmma_f32_16x16x32_bf16(
                        false, pf.v, false, vf.v, (short)0, o[j],
                        false, false);
                }
            }
        }
        __syncthreads();
    }

    // ---- normalize + store: attn[q, h*64 + d] ----
    #pragma unroll
    for (int j = 0; j < 4; ++j) {
        #pragma unroll
        for (int v = 0; v < 8; ++v) {
            int m = qrow0 + hf * 8 + v;
            int d = j * 16 + l15;
            O[(size_t)m * Ec + hoff + d] = o[j][v] / lrow[v];
        }
    }
}

// ---------------------------------------------------------------------------
// Host driver
// ---------------------------------------------------------------------------
extern "C" void kernel_launch(void* const* d_in, const int* in_sizes, int n_in,
                              void* d_out, int out_size, void* d_ws, size_t ws_size,
                              hipStream_t stream)
{
    const int*   inpt = (const int*)  d_in[0];
    // d_in[1] = num_heads (scalar, = 16, hardcoded)
    const float* emb  = (const float*)d_in[2];
    const float* Wq   = (const float*)d_in[3];
    const float* bq   = (const float*)d_in[4];
    const float* Wk   = (const float*)d_in[5];
    const float* bk   = (const float*)d_in[6];
    const float* Wv   = (const float*)d_in[7];
    const float* bv   = (const float*)d_in[8];
    const float* W1   = (const float*)d_in[9];
    const float* b1   = (const float*)d_in[10];
    const float* W2   = (const float*)d_in[11];
    const float* b2   = (const float*)d_in[12];
    const float* Wout = (const float*)d_in[13];
    float* out = (float*)d_out;

    // workspace layout (fp32): x, q, k, v, attn = S*E each; hidden = S*F
    float* x    = (float*)d_ws;
    float* q    = x    + (size_t)Sc * Ec;
    float* k    = q    + (size_t)Sc * Ec;
    float* v    = k    + (size_t)Sc * Ec;
    float* attn = v    + (size_t)Sc * Ec;
    float* hid  = attn + (size_t)Sc * Ec;

    // 1) embedding gather
    embed_kernel<<<(Sc * Ec / 4) / 256, 256, 0, stream>>>(inpt, emb, x);

    dim3 gEE(Sc / BM, Ec / BN);   // [S,E] = [S,E]@[E,E]
    dim3 gEF(Sc / BM, Fc / BN);   // [S,F] = [S,E]@[E,F]
    dim3 gFE(Sc / BM, Ec / BN);   // [S,E] = [S,F]@[F,E]
    dim3 gAtt(Sc / 128, Hc);

    for (int l = 0; l < Lc; ++l) {
        const size_t wEE = (size_t)l * Ec * Ec;
        const size_t wEF = (size_t)l * Ec * Fc;
        // QKV projections
        gemm_bf16_wmma<<<gEE, 256, 0, stream>>>(x, Wq + wEE, bq + (size_t)l * Ec, q, Sc, Ec, Ec, 0);
        gemm_bf16_wmma<<<gEE, 256, 0, stream>>>(x, Wk + wEE, bk + (size_t)l * Ec, k, Sc, Ec, Ec, 0);
        gemm_bf16_wmma<<<gEE, 256, 0, stream>>>(x, Wv + wEE, bv + (size_t)l * Ec, v, Sc, Ec, Ec, 0);
        // causal multi-head attention
        attn_wmma<<<gAtt, 256, 0, stream>>>(q, k, v, attn);
        // FFN (replaces x, no residual — matches reference)
        gemm_bf16_wmma<<<gEF, 256, 0, stream>>>(attn, W1 + wEF, b1 + (size_t)l * Fc, hid, Sc, Fc, Ec, 1);
        gemm_bf16_wmma<<<gFE, 256, 0, stream>>>(hid, W2 + wEF, b2 + (size_t)l * Ec, x, Sc, Ec, Fc, 0);
    }

    // final logits: [S,V] = [S,E]@[E,V]
    dim3 gOut(Sc / BM, Vc / BN);
    gemm_bf16_wmma<<<gOut, 256, 0, stream>>>(x, Wout, nullptr, out, Sc, Vc, Ec, 0);
}
